// VolumeRenderer_72550587564113
// MI455X (gfx1250) — matrix operand order repaired and checked by reading, not compiled, not run on hardware
//
#include <hip/hip_runtime.h>

// Fused NeRF-style volume renderer for gfx1250 (MI455X).
// Key insight: layer-1 pre-activation is affine in depth along a ray:
//   pre[n] = u[n] + d * v[n],  u/v computed once per ray (per-lane, in A-frag layout).
// So h = relu(u + d*v) is built directly in WMMA A-layout registers (f32 fma),
// and only the genuine K=64 contraction (h @ [Ws|Wt]) uses the matrix pipes:
//   2x v_wmma_f32_16x16x32_f16 per 16-sample tile, C-chained (x2 via unroll-2).
// One wave32 = one ray (256 samples). Then a wave-local exclusive scan
// (__shfl_up, warpSize==32) produces transmittance, probs, and the reductions.

#define N_SAMPLES 256
#define WAVES_PER_BLOCK 8
#define FE_SCALE 7.0f

typedef _Float16 v16h __attribute__((ext_vector_type(16)));
typedef float    v8f  __attribute__((ext_vector_type(8)));
typedef float    f4   __attribute__((ext_vector_type(4)));
typedef int      i4   __attribute__((ext_vector_type(4)));

__global__ __launch_bounds__(256) void volrend_wmma(
    const float* __restrict__ ray_start,
    const float* __restrict__ ray_dir,
    const float* __restrict__ depth,
    const float* __restrict__ dist,
    const int*   __restrict__ vidx,
    const float* __restrict__ W1,   // [6,64] row-major
    const float* __restrict__ b1,   // [64]
    const float* __restrict__ Ws,   // [64,1]
    const float* __restrict__ bs,   // [1]
    const float* __restrict__ Wt,   // [64,3] row-major
    const float* __restrict__ bt,   // [3]
    float* __restrict__ out,        // rgb[N,3] | depths[N] | missed[N] | probs[N,256]
    int n_rays)
{
  __shared__ __align__(16) float sg_lds[WAVES_PER_BLOCK][N_SAMPLES][4]; // sigma,tex

  const int  lane = threadIdx.x & 31;
  const int  wave = threadIdx.x >> 5;
  const int  ray  = blockIdx.x * WAVES_PER_BLOCK + wave;
  const int  n    = lane & 15;       // M row (A side) / N column (B,C side)
  const bool hi   = lane >= 16;
  const int  row0 = hi ? 8 : 0;

  // ---------------- GEMM2 weight fragments: W2[64][16] = [Ws | Wt | 0...] ----------------
  // B layout, K-step s covers K = 32s..32s+31: element e <-> K=32s+(e<8?0:16)+(hi?8:0)+(e&7)
  // (same K order as the A fragment).
  v16h bw2[2];
#pragma unroll
  for (int s = 0; s < 2; ++s) {
    v16h b = {};
#pragma unroll
    for (int e = 0; e < 16; ++e) {
      int k = s * 32 + ((e < 8) ? 0 : 16) + (hi ? 8 : 0) + (e & 7);
      float w = (n == 0) ? Ws[k] : ((n < 4) ? Wt[k * 3 + (n - 1)] : 0.0f);
      b[e] = (_Float16)w;
    }
    bw2[s] = b;
  }
  const float bias2 = (n == 0) ? bs[0] : ((n < 4) ? bt[n - 1] : 0.0f);

  const float rsx = ray_start[ray * 3 + 0], rsy = ray_start[ray * 3 + 1], rsz = ray_start[ray * 3 + 2];
  const float rdx = ray_dir[ray * 3 + 0],  rdy = ray_dir[ray * 3 + 1],  rdz = ray_dir[ray * 3 + 2];

  // ---------------- per-ray affine layer-1 coefficients, in A-frag layout ----------------
  // A layout element e of K-step s:  k = 32s + (e<8 ? 0 : 16) + (hi ? 8 : 0) + (e&7),
  // for row M = lane&15.  h[M][k] = relu(u[k] + d_M * v[k]).
  float uu[2][16], vv[2][16];
#pragma unroll
  for (int s = 0; s < 2; ++s) {
#pragma unroll
    for (int e = 0; e < 16; ++e) {
      int k = s * 32 + ((e < 8) ? 0 : 16) + (hi ? 8 : 0) + (e & 7);
      float w0 = W1[0 * 64 + k], w1 = W1[1 * 64 + k], w2 = W1[2 * 64 + k];
      float w3 = W1[3 * 64 + k], w4 = W1[4 * 64 + k], w5 = W1[5 * 64 + k];
      vv[s][e] = rdx * w0 + rdy * w1 + rdz * w2;
      uu[s][e] = rsx * w0 + rsy * w1 + rsz * w2
               + rdx * w3 + rdy * w4 + rdz * w5 + b1[k];
    }
  }

  const float* dep_ray = depth + (size_t)ray * N_SAMPLES;
  float*       sw      = &sg_lds[wave][0][0];

  // ---------------- 16 tiles of 16 samples: h in-registers, GEMM2 on WMMA ----------------
#pragma unroll 2
  for (int t = 0; t < 16; ++t) {
    // depth of this lane's A-row sample (lanes 16-31 mirror 0-15; no divergence)
    float d = dep_ray[t * 16 + n];

    v8f dacc = {};
#pragma unroll
    for (int s = 0; s < 2; ++s) {
      v16h a2;
#pragma unroll
      for (int e = 0; e < 16; ++e)
        a2[e] = (_Float16)fmaxf(uu[s][e] + d * vv[s][e], 0.0f);
      dacc = __builtin_amdgcn_wmma_f32_16x16x32_f16(false, a2, false, bw2[s],
                                                    (short)0, dacc, false, false);
    }

    // D frag: col n in 0..3 holds {sigma, tex.xyz} for samples row0+v of this tile.
    if (n < 4) {
#pragma unroll
      for (int v = 0; v < 8; ++v)
        sw[(t * 16 + row0 + v) * 4 + n] = dacc[v] + bias2;
    }
  }

  // ---------------- per-ray scan + reductions (8 samples per lane) ----------------
  const int    i0       = lane * 8;
  const float* dist_ray = dist + (size_t)ray * N_SAMPLES;
  const int*   idx_ray  = vidx + (size_t)ray * N_SAMPLES;

  f4 dpA = *(const f4*)&dep_ray[i0];
  f4 dpB = *(const f4*)&dep_ray[i0 + 4];
  f4 dsA = *(const f4*)&dist_ray[i0];
  f4 dsB = *(const f4*)&dist_ray[i0 + 4];
  i4 ixA = *(const i4*)&idx_ray[i0];
  i4 ixB = *(const i4*)&idx_ray[i0 + 4];

  float fe[8], dpv[8], txv[8], tyv[8], tzv[8];
  float lsum = 0.0f;
#pragma unroll
  for (int i = 0; i < 8; ++i) {
    f4 sv = *(const f4*)&sw[(i0 + i) * 4];
    float dd = (i < 4) ? dpA[i] : dpB[i - 4];
    float di = (i < 4) ? dsA[i] : dsB[i - 4];
    int   id = (i < 4) ? ixA[i] : ixB[i - 4];
    float f  = fmaxf(sv[0], 0.0f) * di * FE_SCALE;
    f = (id != -1) ? f : 0.0f;          // samples that missed all voxels
    fe[i] = f;
    lsum += f;
    dpv[i] = dd; txv[i] = sv[1]; tyv[i] = sv[2]; tzv[i] = sv[3];
  }

  // exclusive scan of per-lane sums across the wave (wave32!)
  float sc = lsum;
#pragma unroll
  for (int d = 1; d < 32; d <<= 1) {
    float tmp = __shfl_up(sc, d, 32);
    if (lane >= d) sc += tmp;
  }
  float cum = sc - lsum;                 // sum of fe over all earlier samples

  float T = __expf(-cum);                // transmittance entering this lane's segment
  float probsv[8];
  float psum = 0.f, pdep = 0.f, pr = 0.f, pg = 0.f, pb = 0.f;
#pragma unroll
  for (int i = 0; i < 8; ++i) {
    float e = __expf(-fe[i]);
    float p = (1.0f - e) * T;            // a*b of the reference
    T *= e;
    probsv[i] = p;
    psum += p;
    pdep += p * dpv[i];
    pr += p * txv[i]; pg += p * tyv[i]; pb += p * tzv[i];
  }

  float* out_prb = out + (size_t)5 * n_rays + (size_t)ray * N_SAMPLES;
  f4 o0 = {probsv[0], probsv[1], probsv[2], probsv[3]};
  f4 o1 = {probsv[4], probsv[5], probsv[6], probsv[7]};
  *(f4*)&out_prb[i0]     = o0;
  *(f4*)&out_prb[i0 + 4] = o1;

#pragma unroll
  for (int m = 16; m >= 1; m >>= 1) {
    psum += __shfl_xor(psum, m, 32);
    pdep += __shfl_xor(pdep, m, 32);
    pr   += __shfl_xor(pr, m, 32);
    pg   += __shfl_xor(pg, m, 32);
    pb   += __shfl_xor(pb, m, 32);
  }
  if (lane == 0) {
    out[ray * 3 + 0] = pr;
    out[ray * 3 + 1] = pg;
    out[ray * 3 + 2] = pb;
    out[(size_t)3 * n_rays + ray] = pdep;
    out[(size_t)4 * n_rays + ray] = 1.0f - psum;
  }
}

extern "C" void kernel_launch(void* const* d_in, const int* in_sizes, int n_in,
                              void* d_out, int out_size, void* d_ws, size_t ws_size,
                              hipStream_t stream) {
  (void)n_in; (void)out_size; (void)d_ws; (void)ws_size;
  const float* ray_start = (const float*)d_in[0];
  const float* ray_dir   = (const float*)d_in[1];
  const float* depth     = (const float*)d_in[2];
  const float* dist      = (const float*)d_in[3];
  const int*   vidx      = (const int*)d_in[4];
  const float* W1        = (const float*)d_in[5];
  const float* b1        = (const float*)d_in[6];
  const float* Ws        = (const float*)d_in[7];
  const float* bs        = (const float*)d_in[8];
  const float* Wt        = (const float*)d_in[9];
  const float* bt        = (const float*)d_in[10];
  float* out = (float*)d_out;

  const int n_rays = in_sizes[0] / 3;                 // 8192
  dim3 grid(n_rays / WAVES_PER_BLOCK), block(WAVES_PER_BLOCK * 32);
  volrend_wmma<<<grid, block, 0, stream>>>(ray_start, ray_dir, depth, dist, vidx,
                                           W1, b1, Ws, bs, Wt, bt, out, n_rays);
}
